// Segmenter_87299505258749
// MI455X (gfx1250) — compile-verified, matrix operand order checked by reading, not run on hardware
//
#include <hip/hip_runtime.h>
#include <hip/hip_bf16.h>
#include <math.h>

// ---------------- sizes ----------------
#define BNN   6272      // B*N
#define DM    768
#define KPSI  512
#define KH    256       // KPSI/2
#define NB    8         // batch
#define NPIX  784
#define GRID_W 28

typedef __attribute__((ext_vector_type(16))) __bf16  v16bf;
typedef __attribute__((ext_vector_type(8)))  float   v8f;
typedef __attribute__((ext_vector_type(2)))  float   v2f;
typedef __attribute__((ext_vector_type(4)))  unsigned int u32x4;

union BF16x16 { v16bf v; u32x4 q[2]; };

static __device__ __forceinline__ unsigned short f2bf(float x) {
    unsigned int u = __float_as_uint(x);
    unsigned int r = u + 0x7fffu + ((u >> 16) & 1u);
    return (unsigned short)(r >> 16);
}

// ---------------- K1: row-normalize highlevel feature -> bf16 ----------------
__global__ __launch_bounds__(256) void norm_bf16_kernel(const float* __restrict__ hf,
                                                        unsigned short* __restrict__ fbf) {
    const int r = blockIdx.x, t = threadIdx.x;
    const float* row = hf + (size_t)r * DM;
    float x0 = row[t], x1 = row[t + 256], x2 = row[t + 512];
    float s = x0 * x0 + x1 * x1 + x2 * x2;
    for (int o = 16; o; o >>= 1) s += __shfl_xor(s, o);
    __shared__ float red[8];
    if ((t & 31) == 0) red[t >> 5] = s;
    __syncthreads();
    float tot = red[0] + red[1] + red[2] + red[3] + red[4] + red[5] + red[6] + red[7];
    float inv = rsqrtf(tot);
    unsigned short* o = fbf + (size_t)r * DM;
    o[t]       = f2bf(x0 * inv);
    o[t + 256] = f2bf(x1 * inv);
    o[t + 512] = f2bf(x2 * inv);
}

// ---------------- K2: fused bf16-WMMA gram + per-row top-16 ----------------
// 392 blocks of 256 threads (8 waves). Each block owns a 16-row strip; sweeps
// 49 col-tiles of 128; per-wave 16x16 WMMA accumulator; per-lane register
// top-16 heaps merged with wave shfl argmax at the end.
__global__ __launch_bounds__(256) void gram_topk_kernel(const unsigned short* __restrict__ fbf,
                                                        float* __restrict__ topv,
                                                        int*   __restrict__ topi) {
    __shared__ unsigned short Al[16 * DM];     // A strip (16 x 768 bf16)
    __shared__ unsigned short Bl[128 * 64];    // B k-chunk (128 cols x 64 k)
    __shared__ float          Ot[16 * 128];    // output tile

    const int tid = threadIdx.x;
    const int lane = tid & 31, wv = tid >> 5;
    const int half = lane >> 4, m = lane & 15;
    const int rowBase = blockIdx.x * 16;

    // load A strip
    {
        const u32x4* gA = (const u32x4*)(fbf + (size_t)rowBase * DM);
        u32x4* sA = (u32x4*)Al;
        for (int i = tid; i < (16 * DM) / 8; i += 256) sA[i] = gA[i];
    }

    float lv[2][16]; int li[2][16];
#pragma unroll
    for (int s = 0; s < 2; s++)
#pragma unroll
        for (int j = 0; j < 16; j++) { lv[s][j] = -1.0f; li[s][j] = 0; }

    __syncthreads();

    for (int cb = 0; cb < BNN; cb += 128) {
        v8f acc = {0.f, 0.f, 0.f, 0.f, 0.f, 0.f, 0.f, 0.f};
        for (int kc = 0; kc < DM; kc += 64) {
            __syncthreads();                   // previous Bl consumed
            {   // load B chunk: 128 cols x 64 k  (2 threads per col)
                const int col = tid >> 1, seg = tid & 1;
                const u32x4* src = (const u32x4*)(fbf + (size_t)(cb + col) * DM + kc + seg * 32);
                u32x4* dst = (u32x4*)(Bl + col * 64 + seg * 32);
                dst[0] = src[0]; dst[1] = src[1]; dst[2] = src[2]; dst[3] = src[3];
            }
            __syncthreads();
#pragma unroll
            for (int ks = 0; ks < 64; ks += 32) {
                const int k0 = kc + ks;
                BF16x16 a, b;
                a.q[0] = *(const u32x4*)&Al[m * DM + k0 + half * 8];
                a.q[1] = *(const u32x4*)&Al[m * DM + k0 + 16 + half * 8];
                const int bcol = wv * 16 + m;
                b.q[0] = *(const u32x4*)&Bl[bcol * 64 + ks + half * 8];
                b.q[1] = *(const u32x4*)&Bl[bcol * 64 + ks + 16 + half * 8];
                acc = __builtin_amdgcn_wmma_f32_16x16x32_bf16(
                    false, a.v, false, b.v, (short)0, acc, false, false);
            }
        }
        __syncthreads();
#pragma unroll
        for (int r = 0; r < 8; r++) {
            const int M = r + half * 8;
            Ot[M * 128 + wv * 16 + m] = acc[r];
        }
        __syncthreads();
        // top-k update: wave wv owns strip rows 2wv, 2wv+1
#pragma unroll
        for (int s = 0; s < 2; s++) {
            const int rrow = 2 * wv + s;
            const int grow = rowBase + rrow;
#pragma unroll
            for (int t = 0; t < 4; t++) {
                const int c = lane + 32 * t;
                const int gcol = cb + c;
                float v = fmaxf(Ot[rrow * 128 + c], 0.f);
                if (gcol != grow) {
                    float mn = lv[s][0]; int mj = 0;
#pragma unroll
                    for (int j = 1; j < 16; j++) if (lv[s][j] < mn) { mn = lv[s][j]; mj = j; }
                    if (v > mn) {
#pragma unroll
                        for (int j = 0; j < 16; j++) if (j == mj) { lv[s][j] = v; li[s][j] = gcol; }
                    }
                }
            }
        }
        __syncthreads();
    }

    // merge 32 lanes x 16 -> exact top-16, via wave argmax rounds
#pragma unroll
    for (int s = 0; s < 2; s++) {
        const int grow = rowBase + 2 * wv + s;
        for (int round = 0; round < 16; round++) {
            float bv = -3e38f; int bj = 0;
#pragma unroll
            for (int j = 0; j < 16; j++) if (lv[s][j] > bv) { bv = lv[s][j]; bj = j; }
            int bi = 0;
#pragma unroll
            for (int j = 0; j < 16; j++) if (j == bj) bi = li[s][j];
            float mv = bv; int ml = lane;
            for (int o = 16; o; o >>= 1) {
                float ov = __shfl_xor(mv, o); int ol = __shfl_xor(ml, o);
                if (ov > mv || (ov == mv && ol < ml)) { mv = ov; ml = ol; }
            }
            const int widx = __shfl(bi, ml);
            const int wj   = __shfl(bj, ml);
            if (lane == ml) {
#pragma unroll
                for (int j = 0; j < 16; j++) if (j == wj) lv[s][j] = -3e38f;
            }
            if (lane == 0) { topv[grow * 16 + round] = mv; topi[grow * 16 + round] = widx; }
        }
    }
}

// ---------------- K3: symmetric row sums of knn gram ----------------
__global__ __launch_bounds__(256) void knn_rowsum_kernel(const float* __restrict__ topv,
                                                         const int* __restrict__ topi,
                                                         float* __restrict__ rs) {
    const int tid = blockIdx.x * 256 + threadIdx.x;
    if (tid >= BNN * 16) return;
    const int r = tid >> 4;
    const int c = topi[tid];
    const float v = 0.5f * topv[tid];
    atomicAdd(&rs[r], v);
    atomicAdd(&rs[c], v);
}

__global__ __launch_bounds__(256) void rsqrt_kernel(const float* __restrict__ rs,
                                                    float* __restrict__ dinv, int n) {
    const int i = blockIdx.x * 256 + threadIdx.x;
    if (i < n) dinv[i] = (rs[i] > 0.f) ? rsqrtf(rs[i]) : 0.f;
}

// ---------------- K5: GP1 = Gknn @ P1 (sparse, symmetric scatter) ----------------
__global__ __launch_bounds__(256) void knn_gp_kernel(const float* __restrict__ topv,
                                                     const int* __restrict__ topi,
                                                     const float* __restrict__ dinv,
                                                     const float* __restrict__ Psi,
                                                     float* __restrict__ GP) {
    const int r = blockIdx.x, t = threadIdx.x;
    const float dr = dinv[r];
    const float pr = Psi[(size_t)r * KPSI + t];     // P1 column t of row r
    float acc = 0.f;
    for (int j = 0; j < 16; j++) {
        const int c = topi[r * 16 + j];
        const float v = 0.5f * topv[r * 16 + j];
        const float dc = dinv[c];
        acc += v * dc * Psi[(size_t)c * KPSI + t];                 // outgoing
        atomicAdd(&GP[(size_t)c * KH + t], v * dr * dc * pr);      // incoming
    }
    atomicAdd(&GP[(size_t)r * KH + t], dr * acc);
}

// ---------------- pixel features + top-k ----------------
static __device__ __forceinline__ void pixfeat(const float* __restrict__ im, int b, int c,
                                               float dw, float* f) {
    f[0] = ((im[(size_t)(b * 3 + 0) * NPIX + c] + 1.f) * 0.5f) * (1.f / 16.f);
    f[1] = ((im[(size_t)(b * 3 + 1) * NPIX + c] + 1.f) * 0.5f) * (1.f / 16.f);
    f[2] = ((im[(size_t)(b * 3 + 2) * NPIX + c] + 1.f) * 0.5f) * (1.f / 16.f);
    const int cx = c % GRID_W, cy = c / GRID_W;
    f[3] = (cx * (1.f / 27.f)) * dw;
    f[4] = (cy * (1.f / 27.f)) * dw;
}

template <int KN>
static __device__ __forceinline__ void pix_pass(const float* __restrict__ im, int b, int rl,
                                                float dw, int* __restrict__ outIdx) {
    const int lane = threadIdx.x & 31;
    float fr[5]; pixfeat(im, b, rl, dw, fr);
    float lv[KN]; int li[KN];
#pragma unroll
    for (int j = 0; j < KN; j++) { lv[j] = -3e38f; li[j] = 0; }
    for (int t = 0; t < 25; t++) {
        const int c = lane + 32 * t;
        if (c >= NPIX || c == rl) continue;
        float fc[5]; pixfeat(im, b, c, dw, fc);
        float d2 = 0.f;
#pragma unroll
        for (int d = 0; d < 5; d++) { const float e = fr[d] - fc[d]; d2 += e * e; }
        const float key = -d2;          // monotonic with -sqrt(d2)
        float mn = lv[0]; int mj = 0;
#pragma unroll
        for (int j = 1; j < KN; j++) if (lv[j] < mn) { mn = lv[j]; mj = j; }
        if (key > mn) {
#pragma unroll
            for (int j = 0; j < KN; j++) if (j == mj) { lv[j] = key; li[j] = c; }
        }
    }
    for (int round = 0; round < KN; round++) {
        float bv = -3e38f; int bj = 0;
#pragma unroll
        for (int j = 0; j < KN; j++) if (lv[j] > bv) { bv = lv[j]; bj = j; }
        int bi = 0;
#pragma unroll
        for (int j = 0; j < KN; j++) if (j == bj) bi = li[j];
        float mv = bv; int ml = lane;
        for (int o = 16; o; o >>= 1) {
            float ov = __shfl_xor(mv, o); int ol = __shfl_xor(ml, o);
            if (ov > mv || (ov == mv && ol < ml)) { mv = ov; ml = ol; }
        }
        const int widx = __shfl(bi, ml);
        const int wj   = __shfl(bj, ml);
        if (lane == ml) {
#pragma unroll
            for (int j = 0; j < KN; j++) if (j == wj) lv[j] = -3e38f;
        }
        if (lane == 0) outIdx[round] = widx;
    }
}

__global__ __launch_bounds__(32) void pix_topk_kernel(const float* __restrict__ im,
                                                      int* __restrict__ nbr10,
                                                      int* __restrict__ nbr3) {
    const int row = blockIdx.x;
    const int b = row / NPIX, rl = row % NPIX;
    pix_pass<10>(im, b, rl, 2.0f, nbr10 + (size_t)row * 10);
    pix_pass<3>(im, b, rl, 0.1f, nbr3 + (size_t)row * 3);
}

// ---------------- adjacency (dense byte mask, exact boolean union) ----------------
__global__ __launch_bounds__(256) void adj_scatter_kernel(const int* __restrict__ nbr10,
                                                          const int* __restrict__ nbr3,
                                                          unsigned char* __restrict__ adj) {
    const int tid = blockIdx.x * 256 + threadIdx.x;
    if (tid >= NB * NPIX * 13) return;
    const int b = tid / (NPIX * 13);
    const int rem = tid % (NPIX * 13);
    const int r = rem / 13, j = rem % 13;
    const int c = (j < 10) ? nbr10[(size_t)(b * NPIX + r) * 10 + j]
                           : nbr3[(size_t)(b * NPIX + r) * 3 + (j - 10)];
    unsigned char* A = adj + (size_t)b * NPIX * NPIX;
    A[(size_t)r * NPIX + c] = 1;
    A[(size_t)c * NPIX + r] = 1;
}

__global__ __launch_bounds__(32) void adj_rowsum_kernel(const unsigned char* __restrict__ adj,
                                                        float* __restrict__ dinv) {
    const int row = blockIdx.x;
    const int b = row / NPIX, rl = row % NPIX, lane = threadIdx.x;
    const unsigned char* arow = adj + (size_t)b * NPIX * NPIX + (size_t)rl * NPIX;
    int s = 0;
    for (int c = lane; c < NPIX; c += 32) s += arow[c];
    for (int o = 16; o; o >>= 1) s += __shfl_xor(s, o);
    if (lane == 0) dinv[row] = (s > 0) ? rsqrtf((float)s) : 0.f;
}

// ---------------- GP2 = Gpix @ P2 (block-diagonal sparse) ----------------
__global__ __launch_bounds__(256) void pix_gp_kernel(const unsigned char* __restrict__ adj,
                                                     const float* __restrict__ dinv,
                                                     const float* __restrict__ Psi,
                                                     float* __restrict__ GP) {
    const int row = blockIdx.x;
    const int b = row / NPIX, rl = row % NPIX, t = threadIdx.x;
    __shared__ int idxs[NPIX];
    __shared__ int cnt;
    if (t == 0) cnt = 0;
    __syncthreads();
    const unsigned char* arow = adj + (size_t)b * NPIX * NPIX + (size_t)rl * NPIX;
    for (int c = t; c < NPIX; c += 256)
        if (arow[c]) { int p = atomicAdd(&cnt, 1); idxs[p] = c; }
    __syncthreads();
    float acc = 0.f;
    const int n = cnt;
    for (int p = 0; p < n; p++) {
        const int g = b * NPIX + idxs[p];
        acc += dinv[g] * Psi[(size_t)g * KPSI + KH + t];
    }
    GP[(size_t)row * KH + t] = dinv[row] * acc;
}

// ---------------- K10: R = s^2 * P^T (GP) via fp32 WMMA; reduce trace & triu^2 ----
__global__ __launch_bounds__(32) void r_reduce_kernel(const float* __restrict__ Psi,
                                                      const float* __restrict__ GP,
                                                      int colOff, float wgt,
                                                      float* __restrict__ out) {
    const int lane = threadIdx.x;
    const int half = lane >> 4, m = lane & 15;
    const int tileM = (blockIdx.x & 15) * 16;
    const int tileN = ((blockIdx.x >> 4) & 15) * 16;
    v8f acc = {0.f, 0.f, 0.f, 0.f, 0.f, 0.f, 0.f, 0.f};
    for (int k = 0; k < BNN; k += 4) {
        const int ka = k + half * 2;
        v2f a, b;
        a.x = Psi[(size_t)ka * KPSI + colOff + tileM + m];
        a.y = Psi[(size_t)(ka + 1) * KPSI + colOff + tileM + m];
        b.x = GP[(size_t)ka * KH + tileN + m];
        b.y = GP[(size_t)(ka + 1) * KH + tileN + m];
        acc = __builtin_amdgcn_wmma_f32_16x16x4_f32(
            false, a, false, b, (short)0, acc, false, false);
    }
    float tr = 0.f, tq = 0.f;
#pragma unroll
    for (int r = 0; r < 8; r++) {
        const int i = tileM + r + half * 8;
        const int j = tileN + m;
        const float v = acc[r];
        if (i == j) tr += v;
        if (j > i)  tq += v * v;
    }
    for (int o = 16; o; o >>= 1) { tr += __shfl_xor(tr, o); tq += __shfl_xor(tq, o); }
    if (lane == 0) {
        const float s2 = 10.0f / (float)BNN;           // T / BN  (Psi scale squared)
        atomicAdd(&out[0], -wgt * s2 * tr * (1.0f / (float)KPSI));
        atomicAdd(&out[1],  wgt * s2 * s2 * tq * (1.0f / (float)KPSI));  // ALPHA=1
    }
}

// ---------------- host launcher ----------------
extern "C" void kernel_launch(void* const* d_in, const int* in_sizes, int n_in,
                              void* d_out, int out_size, void* d_ws, size_t ws_size,
                              hipStream_t stream) {
    const float* hf  = (const float*)d_in[2];   // highlevel_feature (B,N,D)
    const float* Psi = (const float*)d_in[3];   // (B,N,KPSI)
    const float* im  = (const float*)d_in[4];   // (B,3,H,W)
    float* out = (float*)d_out;                 // [loss, reg]

    char* w = (char*)d_ws;
    unsigned short* fbf  = (unsigned short*)w;          w += (size_t)BNN * DM * 2;
    float* topv          = (float*)w;                   w += (size_t)BNN * 16 * 4;
    int*   topi          = (int*)w;                     w += (size_t)BNN * 16 * 4;
    float* rs1           = (float*)w;                   w += (size_t)BNN * 4;
    float* dinv1         = (float*)w;                   w += (size_t)BNN * 4;
    float* GP1           = (float*)w;                   w += (size_t)BNN * KH * 4;
    int*   nbr10         = (int*)w;                     w += (size_t)NB * NPIX * 10 * 4;
    int*   nbr3          = (int*)w;                     w += (size_t)NB * NPIX * 3 * 4;
    unsigned char* adj   = (unsigned char*)w;           w += (size_t)NB * NPIX * NPIX;
    float* dinv2         = (float*)w;                   w += (size_t)BNN * 4;
    float* GP2           = (float*)w;                   w += (size_t)BNN * KH * 4;

    hipMemsetAsync(out, 0, 2 * sizeof(float), stream);
    hipMemsetAsync(rs1, 0, (size_t)BNN * 4, stream);
    hipMemsetAsync(GP1, 0, (size_t)BNN * KH * 4, stream);
    hipMemsetAsync(adj, 0, (size_t)NB * NPIX * NPIX, stream);

    // KNN gram branch
    norm_bf16_kernel<<<BNN, 256, 0, stream>>>(hf, fbf);
    gram_topk_kernel<<<BNN / 16, 256, 0, stream>>>(fbf, topv, topi);
    knn_rowsum_kernel<<<(BNN * 16) / 256, 256, 0, stream>>>(topv, topi, rs1);
    rsqrt_kernel<<<(BNN + 255) / 256, 256, 0, stream>>>(rs1, dinv1, BNN);
    knn_gp_kernel<<<BNN, 256, 0, stream>>>(topv, topi, dinv1, Psi, GP1);

    // Pixel gram branch
    pix_topk_kernel<<<BNN, 32, 0, stream>>>(im, nbr10, nbr3);
    adj_scatter_kernel<<<(NB * NPIX * 13 + 255) / 256, 256, 0, stream>>>(nbr10, nbr3, adj);
    adj_rowsum_kernel<<<BNN, 32, 0, stream>>>(adj, dinv2);
    pix_gp_kernel<<<BNN, 256, 0, stream>>>(adj, dinv2, Psi, GP2);

    // Final quadratic-form reductions (fp32 WMMA)
    r_reduce_kernel<<<256, 32, 0, stream>>>(Psi, GP1, 0,  1.0f, out);   // W_HIGH
    r_reduce_kernel<<<256, 32, 0, stream>>>(Psi, GP2, KH, 0.1f, out);   // W_MID
}
